// APT_42743514530590
// MI455X (gfx1250) — matrix operand order repaired and checked by reading, not compiled, not run on hardware
//
#include <hip/hip_runtime.h>
#include <hip/hip_bf16.h>
#include <stdint.h>

constexpr int SEQ    = 2048;
constexpr int DIM    = 1024;
constexpr int NH     = 16;
constexpr int HDIM   = 64;
constexpr int NLAYER = 8;
#define NEGV (-1e30f)
#define EPSV (1.1920929e-07f)

typedef __attribute__((ext_vector_type(16))) _Float16 v16h;
typedef __attribute__((ext_vector_type(8)))  _Float16 v8h;
typedef __attribute__((ext_vector_type(4)))  _Float16 v4h;
typedef __attribute__((ext_vector_type(8)))  float    v8f;
typedef __attribute__((ext_vector_type(4)))  unsigned int v4u;
typedef __attribute__((ext_vector_type(8)))  int      v8i;
typedef __attribute__((ext_vector_type(4)))  int      v4i;

union V16U { v16h v; v8h h[2]; };

#if defined(__has_builtin)
#if __has_builtin(__builtin_amdgcn_tensor_load_to_lds) && __has_builtin(__builtin_amdgcn_s_wait_tensorcnt)
#define HAVE_TDM 1
#endif
#endif
#ifndef HAVE_TDM
#define HAVE_TDM 0
#endif

#if __has_include(<hip/amd_detail/amd_gfx1250_TDM.h>)
#define TDM_SIX_ARGS 1
#else
#define TDM_SIX_ARGS 0
#endif

static __device__ __forceinline__ v8f wmma_f16(v16h a, v16h b, v8f c) {
  return __builtin_amdgcn_wmma_f32_16x16x32_f16(false, a, false, b, (short)0, c, false, false);
}

#if HAVE_TDM
// 2-D f16 tile load via Tensor Data Mover: `rows` rows of 32 f16 (64B),
// global row stride = stride_elems, LDS row stride padded to 80B
// (pad_interval=3 -> every 64B, pad_amount=3 -> +16B).  ISA cdna5 ch.8 D# layout.
static __device__ __forceinline__ void tdm_load_tile_f16(
    const _Float16* gsrc, unsigned lds_off, int rows, int stride_elems) {
  unsigned long long ga = (unsigned long long)(uintptr_t)gsrc;
  v4u g0;
  g0[0] = 1u;                                   // count=1, user desc, no gather
  g0[1] = lds_off;                              // lds_addr (bytes)
  g0[2] = (unsigned)ga;                         // global_addr[31:0]
  g0[3] = (unsigned)((ga >> 32) & 0x01ffffffull) | 0x80000000u; // addr[56:32] | type=2
  v8i g1;
  unsigned urows = (unsigned)rows;
  g1[0] = (1 << 16) | (1 << 20) | (3 << 22) | (3 << 25); // data_size=2B, pad_en, int=64B, amt=16B
  g1[1] = (int)(32u << 16);                     // tensor_dim0 = 32 (lo16); atomic addr = 0
  g1[2] = (int)((urows & 0xffffu) << 16);       // tensor_dim0 hi = 0; tensor_dim1 lo = rows
  g1[3] = (int)(32u << 16);                     // tensor_dim1 hi = 0; tile_dim0 = 32
  g1[4] = (int)(urows & 0xffffu);               // tile_dim1 = rows; tile_dim2 = 0
  g1[5] = (int)(unsigned)stride_elems;          // tensor_dim0_stride lo32 (elements)
  g1[6] = 0;
  g1[7] = 0;
  v4i z4 = {};
#if TDM_SIX_ARGS
  v8i z8 = {};
  __builtin_amdgcn_tensor_load_to_lds(g0, g1, z4, z4, z8, 0);
#else
  __builtin_amdgcn_tensor_load_to_lds(g0, g1, z4, z4, 0);
#endif
}
#endif

// ---------------------------------------------------------------------------
// Mask precompute (packed):
// qinfo[s] = samp | cim1<<4
// kinfo[s] = samp | (is0 ? cnt : BIG)<<4     (markov folded into threshold)
// sfirst[v] = first index with sample==v (0 if sample never occurs)
// mask(q,k) = causal && samp_eq && (cim1[q] <= thr[k])
// ---------------------------------------------------------------------------
__global__ void maskprep_kernel(const int* __restrict__ levels,
                                const int* __restrict__ sample_idx,
                                int* kinfo, int* qinfo, int* sfirst) {
  if (threadIdx.x == 0) {
    for (int t = 0; t < 16; ++t) sfirst[t] = 0;
    int seen = 0;
    int c = 0;
    for (int s = 0; s < SEQ; ++s) {
      int i0 = (levels[s] == 0) ? 1 : 0;
      int sm = sample_idx[s] & 15;
      if (!((seen >> sm) & 1)) { sfirst[sm] = s; seen |= (1 << sm); }
      qinfo[s] = sm | (c << 4);
      c += i0;
      int thr = i0 ? c : (1 << 20);
      kinfo[s] = sm | (thr << 4);
    }
  }
}

// ---------------------------------------------------------------------------
// fp32 -> f16 bulk convert
// ---------------------------------------------------------------------------
__global__ __launch_bounds__(256) void cvt_f16_kernel(const float* __restrict__ src,
                                                      _Float16* __restrict__ dst, int n4) {
  int i = blockIdx.x * 256 + threadIdx.x;
  if (i < n4) {
    float4 f = *(const float4*)(src + (long)i * 4);
    v4h h = { (_Float16)f.x, (_Float16)f.y, (_Float16)f.z, (_Float16)f.w };
    *(v4h*)(dst + (long)i * 4) = h;
  }
}

// ---------------------------------------------------------------------------
// Embedding + rmsnorm -> x, x0 (fp32 residual stream)
// ---------------------------------------------------------------------------
__global__ __launch_bounds__(256) void embed_kernel(
    const int* __restrict__ tokens, const int* __restrict__ levels,
    const float* __restrict__ ext, const float* __restrict__ wte,
    const float* __restrict__ lemb, float* __restrict__ x, float* __restrict__ x0) {
  __shared__ float e[DIM];
  __shared__ float red[256];
  int s = blockIdx.x;
  int tok = tokens[s];
  int lv0 = (levels[s] == 0);
  float ss = 0.f;
  for (int d = threadIdx.x; d < DIM; d += 256) {
    float v = lv0 ? ext[(long)s * DIM + d]
                  : (wte[(long)tok * DIM + d] + lemb[d]);
    e[d] = v; ss += v * v;
  }
  red[threadIdx.x] = ss; __syncthreads();
  for (int o = 128; o > 0; o >>= 1) {
    if (threadIdx.x < o) red[threadIdx.x] += red[threadIdx.x + o];
    __syncthreads();
  }
  float inv = rsqrtf(red[0] / (float)DIM + EPSV);
  for (int d = threadIdx.x; d < DIM; d += 256) {
    float v = e[d] * inv;
    x [(long)s * DIM + d] = v;
    x0[(long)s * DIM + d] = v;
  }
}

// ---------------------------------------------------------------------------
// Optional residual mix then rmsnorm; output type templated.
// ---------------------------------------------------------------------------
template<typename OT>
__global__ __launch_bounds__(256) void prenorm_kernel(
    float* __restrict__ x, const float* __restrict__ x0,
    OT* __restrict__ xn, const float* __restrict__ lam2, int mix) {
  __shared__ float e[DIM];
  __shared__ float red[256];
  int s = blockIdx.x;
  float l0 = 1.f, l1 = 0.f;
  if (mix) { l0 = lam2[0]; l1 = lam2[1]; }
  float ss = 0.f;
  for (int d = threadIdx.x; d < DIM; d += 256) {
    float v = x[(long)s * DIM + d];
    if (mix) { v = l0 * v + l1 * x0[(long)s * DIM + d]; x[(long)s * DIM + d] = v; }
    e[d] = v; ss += v * v;
  }
  red[threadIdx.x] = ss; __syncthreads();
  for (int o = 128; o > 0; o >>= 1) {
    if (threadIdx.x < o) red[threadIdx.x] += red[threadIdx.x + o];
    __syncthreads();
  }
  float inv = rsqrtf(red[0] / (float)DIM + EPSV);
  for (int d = threadIdx.x; d < DIM; d += 256) xn[(long)s * DIM + d] = (OT)(e[d] * inv);
}

// ---------------------------------------------------------------------------
// WMMA GEMM: C[M,N] = A[M,K] @ B[N,K]^T, A/B f16 row-major along K.
// Block 256 threads (8 waves), tile 128x128, wave tile 32x64, K chunk 32.
// TDM double-buffered staging (one barrier per chunk) when available.
// MODE 0: store; 1: relu(acc)^2; 2: Cadd + acc.  CT = output element type.
// NOTE: C/Cadd may alias (residual add) -> no restrict on them.
// ---------------------------------------------------------------------------
constexpr int LDH = 40;  // padded LDS row stride in halfs (80B)

template<int MODE, typename CT>
__global__ __launch_bounds__(256) void gemm_wmma(
    const _Float16* __restrict__ A, const _Float16* __restrict__ B,
    CT* C, const float* Cadd, int N, int K) {
  __shared__ _Float16 sm[2][2 * 128 * LDH];   // per buffer: A(128x40) then B(128x40)
  int tid = threadIdx.x, lane = tid & 31, w = tid >> 5;
  int lh = lane & 15, lq = lane >> 4;
  int wm = (w & 3) * 32, wn = (w >> 2) * 64;
  long bm = (long)blockIdx.y * 128, bn = (long)blockIdx.x * 128;
  v8f acc[2][4] = {};
  int nchunks = K / 32;

#if HAVE_TDM
  unsigned ldsA[2], ldsB[2];
  ldsA[0] = (unsigned)(uintptr_t)&sm[0][0];
  ldsB[0] = (unsigned)(uintptr_t)&sm[0][128 * LDH];
  ldsA[1] = (unsigned)(uintptr_t)&sm[1][0];
  ldsB[1] = (unsigned)(uintptr_t)&sm[1][128 * LDH];
  if (w == 0) {
    tdm_load_tile_f16(A + bm * K, ldsA[0], 128, K);
    tdm_load_tile_f16(B + bn * K, ldsB[0], 128, K);
  }
#endif

  for (int c = 0; c < nchunks; ++c) {
    int buf = c & 1;
    _Float16* lA = &sm[buf][0];
    _Float16* lB = &sm[buf][128 * LDH];
#if HAVE_TDM
    if (w == 0) __builtin_amdgcn_s_wait_tensorcnt((short)0);
    __syncthreads();
    if (w == 0 && c + 1 < nchunks) {
      long kc = (long)(c + 1) * 32;
      tdm_load_tile_f16(A + bm * K + kc, ldsA[buf ^ 1], 128, K);
      tdm_load_tile_f16(B + bn * K + kc, ldsB[buf ^ 1], 128, K);
    }
#else
    {
      long kc = (long)c * 32;
      __syncthreads();
      int r = tid >> 1;
      int h16 = (tid & 1) * 16;
      *(v8h*)(&lA[r * LDH + h16])     = *(const v8h*)(A + (bm + r) * K + kc + h16);
      *(v8h*)(&lA[r * LDH + h16 + 8]) = *(const v8h*)(A + (bm + r) * K + kc + h16 + 8);
      *(v8h*)(&lB[r * LDH + h16])     = *(const v8h*)(B + (bn + r) * K + kc + h16);
      *(v8h*)(&lB[r * LDH + h16 + 8]) = *(const v8h*)(B + (bn + r) * K + kc + h16 + 8);
      __syncthreads();
    }
#endif
    V16U af[2];
#pragma unroll
    for (int ms = 0; ms < 2; ++ms) {
      int row = wm + ms * 16 + lh;
      af[ms].h[0] = *(v8h*)(&lA[row * LDH + lq * 8]);       // K 0-7  / 8-15
      af[ms].h[1] = *(v8h*)(&lA[row * LDH + 16 + lq * 8]);  // K 16-23 / 24-31
    }
#pragma unroll
    for (int ns = 0; ns < 4; ++ns) {
      int n = wn + ns * 16 + lh;
      V16U bf;
      bf.h[0] = *(v8h*)(&lB[n * LDH + lq * 8]);
      bf.h[1] = *(v8h*)(&lB[n * LDH + 16 + lq * 8]);
#pragma unroll
      for (int ms = 0; ms < 2; ++ms)
        acc[ms][ns] = wmma_f16(af[ms].v, bf.v, acc[ms][ns]);
    }
  }

#pragma unroll
  for (int ms = 0; ms < 2; ++ms)
#pragma unroll
    for (int ns = 0; ns < 4; ++ns)
#pragma unroll
      for (int r = 0; r < 8; ++r) {
        long row = bm + wm + ms * 16 + lq * 8 + r;
        long col = bn + wn + ns * 16 + lh;
        float val = acc[ms][ns][r];
        if constexpr (MODE == 1) { float t = fmaxf(val, 0.f); val = t * t; }
        if constexpr (MODE == 2) val += Cadd[row * N + col];
        C[row * N + col] = (CT)val;
      }
}

// ---------------------------------------------------------------------------
// v mixing on f16 buffers
// ---------------------------------------------------------------------------
__global__ void vmix_kernel(_Float16* __restrict__ v, _Float16* __restrict__ v1,
                            const float* __restrict__ lamb, int layer) {
  long i = (long)blockIdx.x * 256 + threadIdx.x;
  if (layer == 0) v1[i] = v[i];
  else {
    float l = lamb[0];
    v[i] = (_Float16)((1.f - l) * (float)v[i] + l * (float)v1[i]);
  }
}

// ---------------------------------------------------------------------------
// Per-head transpose v[S][DIM] (head slice 64) -> vt[(h*64+d)][SEQ]
// ---------------------------------------------------------------------------
__global__ __launch_bounds__(256) void vtranspose_kernel(const _Float16* __restrict__ v,
                                                         _Float16* __restrict__ vt) {
  __shared__ _Float16 t[32][72];
  int s0 = blockIdx.x * 32;
  int h  = blockIdx.y;
  int tid = threadIdx.x;
  int r = tid >> 3, c8 = (tid & 7) * 8;
  *(v8h*)(&t[r][c8]) = *(const v8h*)(v + (long)(s0 + r) * DIM + h * HDIM + c8);
  __syncthreads();
  int d = tid & 63, sg = (tid >> 6) * 8;
  v8h o;
#pragma unroll
  for (int j = 0; j < 8; ++j) o[j] = t[sg + j][d];
  *(v8h*)(vt + (long)(h * HDIM + d) * SEQ + s0 + sg) = o;
}

// ---------------------------------------------------------------------------
// Per-(s,h) rmsnorm + rotary on f16 q,k (fp32 math)
// ---------------------------------------------------------------------------
__global__ __launch_bounds__(64) void qknorm_rope_kernel(_Float16* __restrict__ q,
                                                         _Float16* __restrict__ k) {
  __shared__ float buf[HDIM];
  __shared__ float red[HDIM];
  int t = threadIdx.x;
  int srow = blockIdx.x / NH;
  int h    = blockIdx.x % NH;
  long base = (long)srow * DIM + (long)h * HDIM;
  float pos = (float)srow;
  int j = t & 31;
  float invf = __expf(-((float)(2 * j) / (float)HDIM) * 9.210340371976184f);
  float ang = pos * invf;
  float cv = cosf(ang), sv = sinf(ang);
#pragma unroll
  for (int which = 0; which < 2; ++which) {
    _Float16* p = which ? k : q;
    float v = (float)p[base + t];
    red[t] = v * v; __syncthreads();
    for (int o = 32; o > 0; o >>= 1) {
      if (t < o) red[t] += red[t + o];
      __syncthreads();
    }
    float inv = rsqrtf(red[0] / (float)HDIM + EPSV);
    buf[t] = v * inv; __syncthreads();
    float x1 = buf[j], x2 = buf[j + 32];
    float outv = (t < 32) ? (x1 * cv + x2 * sv) : (-x1 * sv + x2 * cv);
    p[base + t] = (_Float16)outv;
    __syncthreads();
  }
}

// ---------------------------------------------------------------------------
// Flash attention: one wave per (32-query block, head), 32-key tiles.
// Max-free softmax (|score*scale| <= 8 since q/k are rmsnorm'd) ->
// per-lane partial row sums, single cross-lane reduction after the loop.
// Sample-skip: keys before first token of this block's sample are masked for
// every query in the block -> start loop at sfirst[samp[q0]] & ~31.
// ---------------------------------------------------------------------------
__global__ __launch_bounds__(32) void attn_kernel(
    const _Float16* __restrict__ q, const _Float16* __restrict__ k,
    const _Float16* __restrict__ vt, _Float16* __restrict__ y,
    const int* __restrict__ kinfo, const int* __restrict__ qinfo,
    const int* __restrict__ sfirst) {
  __shared__ _Float16 lP[32 * 32];
  int lane = threadIdx.x;
  int qblk = blockIdx.x, h = blockIdx.y;
  int lh = lane & 15, lq = lane >> 4;
  int q0 = qblk * 32;
  long hoff = (long)h * HDIM;
  const float scale = 0.125f;

  V16U qf[2][2];
#pragma unroll
  for (int qs = 0; qs < 2; ++qs) {
    const _Float16* qp = q + (long)(q0 + qs * 16 + lh) * DIM + hoff;
#pragma unroll
    for (int f = 0; f < 2; ++f) {
      qf[qs][f].h[0] = *(const v8h*)(qp + f * 32 + lq * 8);
      qf[qs][f].h[1] = *(const v8h*)(qp + f * 32 + 16 + lq * 8);
    }
  }
  int qw[2][8];
#pragma unroll
  for (int qs = 0; qs < 2; ++qs)
#pragma unroll
    for (int r = 0; r < 8; ++r) qw[qs][r] = qinfo[q0 + qs * 16 + lq * 8 + r];

  v8f yacc[2][4] = {};
  float srun[2][8];
#pragma unroll
  for (int qs = 0; qs < 2; ++qs)
#pragma unroll
    for (int r = 0; r < 8; ++r) srun[qs][r] = 0.f;

  auto do_tile = [&](int ks, bool causal) {
    V16U kb[2][2];
#pragma unroll
    for (int tile = 0; tile < 2; ++tile) {
      const _Float16* kp = k + (long)(ks + tile * 16 + lh) * DIM + hoff;
      kb[tile][0].h[0] = *(const v8h*)(kp + lq * 8);
      kb[tile][0].h[1] = *(const v8h*)(kp + 16 + lq * 8);
      kb[tile][1].h[0] = *(const v8h*)(kp + 32 + lq * 8);
      kb[tile][1].h[1] = *(const v8h*)(kp + 32 + 16 + lq * 8);
    }
    v8f sc[2][2];
#pragma unroll
    for (int qs = 0; qs < 2; ++qs)
#pragma unroll
      for (int tile = 0; tile < 2; ++tile) {
        v8f s = {};
        s = wmma_f16(qf[qs][0].v, kb[tile][0].v, s);
        s = wmma_f16(qf[qs][1].v, kb[tile][1].v, s);
        sc[qs][tile] = s;
      }

    int kn0 = ks + lh, kn1 = ks + 16 + lh;
    int ki0 = kinfo[kn0], ki1 = kinfo[kn1];
    int sk0 = ki0 & 15, ck0 = ki0 >> 4;
    int sk1 = ki1 & 15, ck1 = ki1 >> 4;
#pragma unroll
    for (int qs = 0; qs < 2; ++qs)
#pragma unroll
      for (int r = 0; r < 8; ++r) {
        int sq = qw[qs][r] & 15, cq = qw[qs][r] >> 4;
        bool ok0 = (sq == sk0) && (cq <= ck0);
        bool ok1 = (sq == sk1) && (cq <= ck1);
        if (causal) {
          int qi = q0 + qs * 16 + lq * 8 + r;
          ok0 = ok0 && (qi >= kn0);
          ok1 = ok1 && (qi >= kn1);
        }
        float p0 = ok0 ? __expf(sc[qs][0][r] * scale) : 0.f;
        float p1 = ok1 ? __expf(sc[qs][1][r] * scale) : 0.f;
        sc[qs][0][r] = p0; sc[qs][1][r] = p1;
        srun[qs][r] += p0 + p1;          // per-lane partial; reduced once at end
      }

    __syncthreads();
#pragma unroll
    for (int qs = 0; qs < 2; ++qs)
#pragma unroll
      for (int r = 0; r < 8; ++r) {
        int m = qs * 16 + lq * 8 + r;
        lP[m * 32 + lh]      = (_Float16)sc[qs][0][r];
        lP[m * 32 + 16 + lh] = (_Float16)sc[qs][1][r];
      }
    __syncthreads();
    V16U pf[2];
#pragma unroll
    for (int qs = 0; qs < 2; ++qs) {
      pf[qs].h[0] = *(v8h*)(&lP[(qs * 16 + lh) * 32 + lq * 8]);
      pf[qs].h[1] = *(v8h*)(&lP[(qs * 16 + lh) * 32 + 16 + lq * 8]);
    }
#pragma unroll
    for (int a = 0; a < 4; ++a) {
      const _Float16* vtp = vt + (hoff + a * 16 + lh) * SEQ + ks;
      V16U vf;
      vf.h[0] = *(const v8h*)(vtp + lq * 8);
      vf.h[1] = *(const v8h*)(vtp + 16 + lq * 8);
      yacc[0][a] = wmma_f16(pf[0].v, vf.v, yacc[0][a]);
      yacc[1][a] = wmma_f16(pf[1].v, vf.v, yacc[1][a]);
    }
  };

  int ks0 = sfirst[qinfo[q0] & 15] & ~31;
#pragma unroll 1
  for (int ks = ks0; ks < q0; ks += 32) do_tile(ks, false);
  do_tile(q0, true);

  // one cross-lane reduction of the partial row sums
#pragma unroll
  for (int qs = 0; qs < 2; ++qs)
#pragma unroll
    for (int r = 0; r < 8; ++r) {
      float st = srun[qs][r];
#pragma unroll
      for (int o = 8; o > 0; o >>= 1) st += __shfl_xor(st, o, 32);
      srun[qs][r] = st;
    }

#pragma unroll
  for (int qs = 0; qs < 2; ++qs)
#pragma unroll
    for (int a = 0; a < 4; ++a)
#pragma unroll
      for (int r = 0; r < 8; ++r) {
        long row = q0 + qs * 16 + lq * 8 + r;
        long col = (long)h * HDIM + a * 16 + lh;
        y[row * DIM + col] = (_Float16)(yacc[qs][a][r] / srun[qs][r]);
      }
}

// ---------------------------------------------------------------------------
extern "C" void kernel_launch(void* const* d_in, const int* in_sizes, int n_in,
                              void* d_out, int out_size, void* d_ws, size_t ws_size,
                              hipStream_t stream) {
  (void)in_sizes; (void)n_in; (void)out_size; (void)ws_size;
  const int*   tokens     = (const int*)d_in[0];
  const int*   levels     = (const int*)d_in[1];
  const int*   sample_idx = (const int*)d_in[2];
  const float* ext        = (const float*)d_in[3];
  const float* wte        = (const float*)d_in[4];
  const float* lemb       = (const float*)d_in[5];
  const float* Wq         = (const float*)d_in[6];
  const float* Wk         = (const float*)d_in[7];
  const float* Wv         = (const float*)d_in[8];
  const float* Wo         = (const float*)d_in[9];
  const float* lamb       = (const float*)d_in[10];
  const float* lambdas    = (const float*)d_in[11];
  const float* W1         = (const float*)d_in[12];
  const float* W2         = (const float*)d_in[13];
  float* out = (float*)d_out;

  const size_t SD = (size_t)SEQ * DIM;
  const size_t MM = (size_t)DIM * DIM;
  char* p = (char*)d_ws;
  auto alloc = [&](size_t bytes) { char* r = p; p += (bytes + 255) & ~(size_t)255; return r; };
  float*    x   = (float*)   alloc(SD * 4);
  float*    x0  = (float*)   alloc(SD * 4);
  _Float16* xnh = (_Float16*)alloc(SD * 2);
  _Float16* qh  = (_Float16*)alloc(SD * 2);
  _Float16* kh  = (_Float16*)alloc(SD * 2);
  _Float16* vh  = (_Float16*)alloc(SD * 2);
  _Float16* v1h = (_Float16*)alloc(SD * 2);
  _Float16* vth = (_Float16*)alloc(SD * 2);
  _Float16* yh  = (_Float16*)alloc(SD * 2);
  _Float16* hh  = (_Float16*)alloc(4 * SD * 2);
  _Float16* wqh = (_Float16*)alloc(MM * 2);
  _Float16* wkh = (_Float16*)alloc(MM * 2);
  _Float16* wvh = (_Float16*)alloc(MM * 2);
  _Float16* woh = (_Float16*)alloc(MM * 2);
  _Float16* w1h = (_Float16*)alloc(4 * MM * 2);
  _Float16* w2h = (_Float16*)alloc(4 * MM * 2);
  int* kinfo  = (int*)alloc(SEQ * 4);
  int* qinfo  = (int*)alloc(SEQ * 4);
  int* sfirst = (int*)alloc(16 * 4);

  maskprep_kernel<<<1, 32, 0, stream>>>(levels, sample_idx, kinfo, qinfo, sfirst);
  embed_kernel<<<SEQ, 256, 0, stream>>>(tokens, levels, ext, wte, lemb, x, x0);

  dim3 gP(DIM / 128, SEQ / 128);       // N=1024
  dim3 gH(4 * DIM / 128, SEQ / 128);   // N=4096
  const int cv1 = (int)(MM / 4), cv4 = (int)(MM);

  for (int i = 0; i < NLAYER; ++i) {
    cvt_f16_kernel<<<cv1 / 256, 256, 0, stream>>>(Wq + i * MM, wqh, cv1);
    cvt_f16_kernel<<<cv1 / 256, 256, 0, stream>>>(Wk + i * MM, wkh, cv1);
    cvt_f16_kernel<<<cv1 / 256, 256, 0, stream>>>(Wv + i * MM, wvh, cv1);
    cvt_f16_kernel<<<cv1 / 256, 256, 0, stream>>>(Wo + i * MM, woh, cv1);
    cvt_f16_kernel<<<cv4 / 256, 256, 0, stream>>>(W1 + i * 4 * MM, w1h, cv4);
    cvt_f16_kernel<<<cv4 / 256, 256, 0, stream>>>(W2 + i * 4 * MM, w2h, cv4);

    prenorm_kernel<_Float16><<<SEQ, 256, 0, stream>>>(x, x0, xnh, lambdas + 2 * i, 1);
    gemm_wmma<0, _Float16><<<gP, 256, 0, stream>>>(xnh, wqh, qh, nullptr, DIM, DIM);
    gemm_wmma<0, _Float16><<<gP, 256, 0, stream>>>(xnh, wkh, kh, nullptr, DIM, DIM);
    gemm_wmma<0, _Float16><<<gP, 256, 0, stream>>>(xnh, wvh, vh, nullptr, DIM, DIM);
    vmix_kernel<<<(int)(SD / 256), 256, 0, stream>>>(vh, v1h, lamb + i, i);
    vtranspose_kernel<<<dim3(SEQ / 32, NH), 256, 0, stream>>>(vh, vth);
    qknorm_rope_kernel<<<SEQ * NH, 64, 0, stream>>>(qh, kh);
    attn_kernel<<<dim3(SEQ / 32, NH), 32, 0, stream>>>(qh, kh, vth, yh, kinfo, qinfo, sfirst);
    gemm_wmma<2, float><<<gP, 256, 0, stream>>>(yh, woh, x, x, DIM, DIM);
    prenorm_kernel<_Float16><<<SEQ, 256, 0, stream>>>(x, nullptr, xnh, nullptr, 0);
    gemm_wmma<1, _Float16><<<gH, 256, 0, stream>>>(xnh, w1h, hh, nullptr, 4 * DIM, DIM);
    gemm_wmma<2, float><<<gP, 256, 0, stream>>>(hh, w2h, x, x, DIM, 4 * DIM);
  }
  prenorm_kernel<float><<<SEQ, 256, 0, stream>>>(x, nullptr, out, nullptr, 0);
}
